// myVMLSTM_21998822490488
// MI455X (gfx1250) — matrix operand order repaired and checked by reading, not compiled, not run on hardware
//
#include <hip/hip_runtime.h>
#include <hip/hip_bf16.h>
#include <math.h>

// ---------------------------------------------------------------------------
// VMLSTM on gfx1250: bf16 WMMA GEMMs + TDM (tensor_load_to_lds) A-tile staging.
//   Precompute:  P = x @ u_x                       (time-parallel big GEMM)
//   Per step t:  Q = h @ u_h                       (WMMA, 64x256x1024)
//                gates = [P_t | Q] @ [w_x ; w_h]^T + x*alpha_x + h*alpha_h + b
//                LSTM pointwise update (f32), state in-place in d_out slots.
// A tiles are contiguous in memory -> one 1-D TDM descriptor per tile,
// issued by wave 0, synced with s_wait_tensorcnt + workgroup barrier; all
// waves then read A fragments from LDS (ds_load_b128), B streams from L2.
// ---------------------------------------------------------------------------

static constexpr int LT = 256;   // T
static constexpr int LB = 64;    // B
static constexpr int LD = 1024;  // D
static constexpr int LH = 1024;  // H
static constexpr int LR = 256;   // R

typedef __attribute__((ext_vector_type(16))) __bf16   v16bf;
typedef __attribute__((ext_vector_type(8)))  __bf16   v8bf;
typedef __attribute__((ext_vector_type(8)))  float    v8f;
typedef __attribute__((ext_vector_type(4)))  unsigned v4u;
typedef __attribute__((ext_vector_type(8)))  int      v8i;
typedef __attribute__((ext_vector_type(4)))  int      v4i;

// float -> bf16, round-to-nearest-even, via bit ops
__device__ __forceinline__ __bf16 f2bf(float f) {
  unsigned u = __builtin_bit_cast(unsigned, f);
  unsigned r = u + 0x7FFFu + ((u >> 16) & 1u);
  return __builtin_bit_cast(__bf16, (unsigned short)(r >> 16));
}

// Flat LDS pointer -> LDS byte offset (ISA: LDS aperture uses addr[31:0])
__device__ __forceinline__ unsigned lds_byte_off(const void* p) {
  return (unsigned)(unsigned long long)p;
}

// ---------------------------------------------------------------------------
// Tensor Data Mover: 1-D contiguous copy global -> LDS.
// D# per CDNA5 ISA: group0 = {flags, lds_addr,
// global_addr[31:0], global_addr[56:32] | type=2}; group1: data_size=4B,
// tensor_dim0 = tile_dim0 = n (single row, tile_dim1/2 unused), dim0 stride=n.
// This toolchain exposes the 6-arg builtin:
//   (uint32x4 g0, int32x8 g1, int32x4 g2, int32x4 g3, int32x8 g4, i32 cpol)
// count=1 in g0 -> only the first descriptor is live; extra groups zeroed.
// nbytes must be a multiple of 4 and < 256 KB.
// ---------------------------------------------------------------------------
__device__ __forceinline__ void tdm_load_1d(unsigned lds_addr, const void* src,
                                            unsigned nbytes) {
  unsigned long long ga = (unsigned long long)src;
  unsigned n = nbytes >> 2;                          // 4-byte elements
  v4u g0 = { 1u,                                     // count=1, user descriptor
             lds_addr,                               // lds_addr (bytes)
             (unsigned)ga,                           // global_addr[31:0]
             ((unsigned)(ga >> 32) & 0x01FFFFFFu) | (2u << 30) };  // [56:32]|type=2
  v8i g1 = { (int)(2u << 16),                        // wg_mask=0, data_size=4B
             (int)((n & 0xFFFFu) << 16),             // tensor_dim0[15:0]
             (int)(((n >> 16) & 0xFFFFu) | (1u << 16)), // dim0[31:16], tensor_dim1=1
             (int)((n & 0xFFFFu) << 16),             // tile_dim0 = n
             0,                                      // tile_dim1=0, tile_dim2=0
             (int)n, 0, 0 };                         // tensor_dim0_stride = n
  v4i gz4 = { 0, 0, 0, 0 };
  v8i gz8 = { 0, 0, 0, 0, 0, 0, 0, 0 };
  __builtin_amdgcn_tensor_load_to_lds(g0, g1, gz4, gz4, gz8, 0);
}

// A fragment (16x32 bf16) from LDS (row-major tile, lda elements per row).
// ISA A-layout: lane<16: row m=lane, K={0..7,16..23}; lane>=16: K={8..15,24..31}
__device__ __forceinline__ v16bf load_a_frag(const __bf16* A, int lda, int lane) {
  const __bf16* p = A + (size_t)(lane & 15) * lda + ((lane >> 4) << 3);
  v8bf lo = *(const v8bf*)p;
  v8bf hi = *(const v8bf*)(p + 16);
  return __builtin_shufflevector(lo, hi, 0,1,2,3,4,5,6,7,8,9,10,11,12,13,14,15);
}

// B fragment (32x16 bf16) from global; row n of source holds K contiguous.
// ISA B-layout: lanes 0-15 K={0..7,16..23} of col n=lane; lanes 16-31 K={8..15,24..31}.
__device__ __forceinline__ v16bf load_b_frag(const __bf16* Bt, int ldb, int lane) {
  const __bf16* p = Bt + (size_t)(lane & 15) * ldb + ((lane >> 4) << 3);
  v8bf lo = *(const v8bf*)p;
  v8bf hi = *(const v8bf*)(p + 16);
  return __builtin_shufflevector(lo, hi, 0,1,2,3,4,5,6,7,8,9,10,11,12,13,14,15);
}

__device__ __forceinline__ v8f wmma_bf16(v16bf a, v16bf b, v8f c) {
  return __builtin_amdgcn_wmma_f32_16x16x32_bf16(false, a, false, b,
                                                 (short)0, c, false, false);
}

// ---------------------------------------------------------------------------
// Prep kernels
// ---------------------------------------------------------------------------
__global__ void k_cvt_bf16(const float* __restrict__ s, __bf16* __restrict__ d, int n) {
  int i = blockIdx.x * blockDim.x + threadIdx.x;
  if (i < n) d[i] = f2bf(s[i]);
}

__global__ void k_tcvt_bf16(const float* __restrict__ s, __bf16* __restrict__ d,
                            int rows, int cols) {
  int i = blockIdx.x * blockDim.x + threadIdx.x;
  if (i < rows * cols) {
    int r = i / cols, c = i - r * cols;
    d[(size_t)c * rows + r] = f2bf(s[i]);
  }
}

__global__ void k_alpha(const float* __restrict__ ux, const float* __restrict__ uh,
                        const float* __restrict__ wx, const float* __restrict__ wh,
                        const float* __restrict__ bx, const float* __restrict__ bh,
                        const float* __restrict__ dx, const float* __restrict__ dh,
                        float* __restrict__ ax, float* __restrict__ ah,
                        float* __restrict__ bs) {
  int j = blockIdx.x * blockDim.x + threadIdx.x;   // 0..4H-1
  if (j >= 4 * LH) return;
  int d = j & (LH - 1);
  float cx = 0.f, ch = 0.f;
  for (int r = 0; r < LR; ++r) {
    cx += ux[(size_t)d * LR + r] * wx[(size_t)j * LR + r];
    ch += uh[(size_t)d * LR + r] * wh[(size_t)j * LR + r];
  }
  ax[j] = dx[d] - cx;
  ah[j] = dh[d] - ch;
  bs[j] = bx[j] + bh[j];
}

__global__ void k_init_state(const float* __restrict__ h0, const float* __restrict__ c0,
                             float* __restrict__ hf, float* __restrict__ cf,
                             __bf16* __restrict__ hb, int n) {
  int i = blockIdx.x * blockDim.x + threadIdx.x;
  if (i < n) { hf[i] = h0[i]; cf[i] = c0[i]; hb[i] = f2bf(h0[i]); }
}

// ---------------------------------------------------------------------------
// Rank-projection GEMM:  Out[M,256] = A[M,1024] @ B^T   (B^T rows contiguous)
// Block = 4 waves sharing one A tile (same mt, nt = 4 consecutive R-tiles).
// A tile (16x1024 bf16 = 32 KB, contiguous) staged via one TDM descriptor.
// Used for P = x@u_x (grid 4096) and per-step Q = h@u_h (grid 16).
// ---------------------------------------------------------------------------
__global__ void k_gemm_rk(const __bf16* __restrict__ A, const __bf16* __restrict__ Bt,
                          __bf16* __restrict__ Out) {
  __shared__ __bf16 As[16 * 1024];                 // 32 KB A tile
  int wave = threadIdx.x >> 5, lane = threadIdx.x & 31;
  int tile = blockIdx.x * 4 + wave;                // nt varies within block
  int nt = tile & 15;
  int mt = tile >> 4;                              // block-uniform
  if (threadIdx.x < 32) {
    tdm_load_1d(lds_byte_off(As), A + (size_t)(mt * 16) * 1024, 16 * 1024 * 2);
    __builtin_amdgcn_s_wait_tensorcnt(0);
  }
  __syncthreads();

  const __bf16* Brow = Bt + (size_t)(nt * 16) * 1024;
  v8f acc = {};
#pragma unroll 4
  for (int k = 0; k < 1024; k += 32) {
    __builtin_prefetch(Brow + k + 128, 0, 1);
    v16bf a = load_a_frag(As + k, 1024, lane);     // ds_load_b128 from LDS
    v16bf b = load_b_frag(Brow + k, 1024, lane);   // global_load_b128 (L2)
    acc = wmma_bf16(a, b, acc);
  }
  int n = lane & 15, mb = (lane >> 4) * 8;
  __bf16* o = Out + (size_t)(mt * 16 + mb) * LR + nt * 16 + n;
#pragma unroll
  for (int j = 0; j < 8; ++j) o[(size_t)j * LR] = f2bf(acc[j]);
}

// ---------------------------------------------------------------------------
// Per-step fused gates + LSTM update.
// Grid: 4 batch-tiles x 64 d-tiles = 256 blocks of 128 threads (4 waves).
// Wave g computes gate-g 16x16 tile over K = 256(P,w_x) + 256(Q,w_h).
// The two shared A tiles (P-tile, Q-tile; 8 KB each, contiguous) are staged
// once per block via TDM; gate tiles staged in LDS for the pointwise update.
// ---------------------------------------------------------------------------
__global__ void k_step(const __bf16* __restrict__ Pb, const __bf16* __restrict__ Qb,
                       const __bf16* __restrict__ Wx, const __bf16* __restrict__ Wh,
                       const float* __restrict__ ax, const float* __restrict__ ah,
                       const float* __restrict__ bs, const float* __restrict__ x,
                       float* __restrict__ out, float* __restrict__ hf,
                       float* __restrict__ cf, __bf16* __restrict__ hb, int t) {
  __shared__ __bf16 As[2][16 * 256];               // 8 KB P tile + 8 KB Q tile
  __shared__ float gl[4][16][16];
  int wave = threadIdx.x >> 5, lane = threadIdx.x & 31;
  int bt = blockIdx.x >> 6;                        // batch tile 0..3
  int dt = blockIdx.x & 63;                        // hidden-col tile 0..63

  if (threadIdx.x < 32) {
    tdm_load_1d(lds_byte_off(As[0]), Pb + (size_t)(t * LB + bt * 16) * LR, 16 * LR * 2);
    tdm_load_1d(lds_byte_off(As[1]), Qb + (size_t)(bt * 16) * LR, 16 * LR * 2);
    __builtin_amdgcn_s_wait_tensorcnt(0);
  }
  __syncthreads();

  // Input path: A = P tile (LDS), B = w_x rows for gate `wave`
  const __bf16* Brow = Wx + (size_t)(wave * LH + dt * 16) * LR;
  v8f acc = {};
#pragma unroll
  for (int k = 0; k < LR; k += 32) {
    v16bf a = load_a_frag(As[0] + k, LR, lane);
    v16bf b = load_b_frag(Brow + k, LR, lane);
    acc = wmma_bf16(a, b, acc);
  }
  // Hidden path: A = Q tile (LDS), B = w_h rows
  const __bf16* Brow2 = Wh + (size_t)(wave * LH + dt * 16) * LR;
#pragma unroll
  for (int k = 0; k < LR; k += 32) {
    v16bf a = load_a_frag(As[1] + k, LR, lane);
    v16bf b = load_b_frag(Brow2 + k, LR, lane);
    acc = wmma_bf16(a, b, acc);
  }
  // Stage gate tile (C/D layout: vgpr j, lane L -> m=j+8*(L>>4), n=L&15)
  {
    int n = lane & 15, mb = (lane >> 4) * 8;
#pragma unroll
    for (int j = 0; j < 8; ++j) gl[wave][mb + j][n] = acc[j];
  }
  __syncthreads();

  // Pointwise LSTM: 256 elements, 128 threads -> 2 each
  for (int e = threadIdx.x; e < 256; e += 128) {
    int m = e >> 4, n = e & 15;
    int b = bt * 16 + m;
    int d = dt * 16 + n;
    size_t bd = (size_t)b * LH + d;
    float xv = x[(size_t)t * LB * LD + (size_t)b * LD + d];
    float hv = hf[bd];
    float g[4];
#pragma unroll
    for (int gg = 0; gg < 4; ++gg) {
      int col = gg * LH + d;
      g[gg] = gl[gg][m][n] + xv * ax[col] + hv * ah[col] + bs[col];
    }
    float gi = 1.f / (1.f + __expf(-g[0]));
    float gf = 1.f / (1.f + __expf(-g[1]));
    float go = 1.f / (1.f + __expf(-g[2]));
    float gn = tanhf(g[3]);
    float cv = gf * cf[bd] + gi * gn;
    float hn = go * tanhf(cv);
    cf[bd] = cv;
    hf[bd] = hn;
    hb[bd] = f2bf(hn);
    out[(size_t)t * LB * LH + bd] = hn;
  }
}

// ---------------------------------------------------------------------------
extern "C" void kernel_launch(void* const* d_in, const int* in_sizes, int n_in,
                              void* d_out, int out_size, void* d_ws, size_t ws_size,
                              hipStream_t stream) {
  (void)in_sizes; (void)n_in; (void)out_size; (void)ws_size;
  const float* x     = (const float*)d_in[0];
  const float* h0    = (const float*)d_in[1];
  const float* c0    = (const float*)d_in[2];
  const float* u_x   = (const float*)d_in[3];
  const float* u_h   = (const float*)d_in[4];
  const float* w_x   = (const float*)d_in[5];
  const float* w_h   = (const float*)d_in[6];
  const float* b_x   = (const float*)d_in[7];
  const float* b_h   = (const float*)d_in[8];
  const float* dia_x = (const float*)d_in[9];
  const float* dia_h = (const float*)d_in[10];

  float* out = (float*)d_out;                     // [T,B,H]
  float* hf  = out + (size_t)LT * LB * LH;        // running h == final h_f
  float* cf  = hf + (size_t)LB * LH;              // running c == final c_f

  // Workspace carve (deterministic, 256B aligned)
  char* wp = (char*)d_ws;
  auto carve = [&](size_t bytes) -> char* {
    char* p = wp; wp += (bytes + 255) & ~(size_t)255; return p;
  };
  __bf16* xb  = (__bf16*)carve((size_t)LT * LB * LD * 2); // 32 MB
  __bf16* uxT = (__bf16*)carve((size_t)LR * LD * 2);      // u_x^T bf16 [R,D]
  __bf16* uhT = (__bf16*)carve((size_t)LR * LH * 2);      // u_h^T bf16 [R,H]
  __bf16* wxb = (__bf16*)carve((size_t)4 * LH * LR * 2);  // w_x bf16 [4H,R]
  __bf16* whb = (__bf16*)carve((size_t)4 * LH * LR * 2);  // w_h bf16 [4H,R]
  __bf16* Pb  = (__bf16*)carve((size_t)LT * LB * LR * 2); // 8 MB
  __bf16* Qb  = (__bf16*)carve((size_t)LB * LR * 2);
  __bf16* hb  = (__bf16*)carve((size_t)LB * LH * 2);
  float*  ax  = (float*)carve((size_t)4 * LH * 4);
  float*  ah  = (float*)carve((size_t)4 * LH * 4);
  float*  bs  = (float*)carve((size_t)4 * LH * 4);

  // ---- prep ----
  {
    int n = LT * LB * LD;
    k_cvt_bf16<<<n / 256, 256, 0, stream>>>(x, xb, n);
  }
  {
    int n = 4 * LH * LR;
    k_cvt_bf16<<<n / 256, 256, 0, stream>>>(w_x, wxb, n);
    k_cvt_bf16<<<n / 256, 256, 0, stream>>>(w_h, whb, n);
  }
  k_tcvt_bf16<<<(LD * LR) / 256, 256, 0, stream>>>(u_x, uxT, LD, LR);
  k_tcvt_bf16<<<(LH * LR) / 256, 256, 0, stream>>>(u_h, uhT, LH, LR);
  k_alpha<<<(4 * LH) / 256, 256, 0, stream>>>(u_x, u_h, w_x, w_h, b_x, b_h,
                                              dia_x, dia_h, ax, ah, bs);
  k_init_state<<<(LB * LH) / 256, 256, 0, stream>>>(h0, c0, hf, cf, hb, LB * LH);

  // ---- time-parallel input projection: P = x @ u_x ----
  k_gemm_rk<<<(LT * LB / 16) * (LR / 16) / 4, 128, 0, stream>>>(xb, uxT, Pb);

  // ---- serial recurrence ----
  for (int t = 0; t < LT; ++t) {
    k_gemm_rk<<<16, 128, 0, stream>>>(hb, uhT, Qb);
    k_step<<<256, 128, 0, stream>>>(Pb, Qb, wxb, whb, ax, ah, bs, x,
                                    out, hf, cf, hb, t);
  }
}
